// FactorizationMachine_19430432047273
// MI455X (gfx1250) — matrix operand order repaired and checked by reading, not compiled, not run on hardware
//
#include <hip/hip_runtime.h>

// ---------------------------------------------------------------------------
// Factorization Machine forward, MI455X (gfx1250, wave32, WMMA bf16 path).
//
//   out_i = sigmoid( x_i.w + b + 0.5/K * ( sum_k (x_i.k_k)^2  -  x_i^2 . s ) )
//   with s[f] = sum_k kernel[f,k]^2   (collapses the second GEMM)
//
// GEMM x@kernel done as split-bf16 (hi/lo) with fp32 WMMA accumulation:
//   y ~= hi*hi + hi*lo + lo*hi   (3x v_wmma_f32_16x16x32_bf16 per K=32 chunk)
// ---------------------------------------------------------------------------

typedef __attribute__((ext_vector_type(16))) __bf16 v16bf;
typedef __attribute__((ext_vector_type(8)))  float  v8f;

constexpr int Bn      = 16384;
constexpr int Fdim    = 2048;
constexpr int Kdim    = 128;
constexpr int CH      = 32;              // WMMA reduction chunk (bf16 K)
constexpr int NCHUNK  = Fdim / CH;       // 64
constexpr int NT      = Kdim / 16;       // 8 output tiles per row stripe
constexpr int WG_THREADS  = 256;         // 8 wave32s
constexpr int ROWS_PER_WG = (WG_THREADS / 32) * 16;  // 128

// workspace layout (bytes)
constexpr size_t WS_S_OFF   = 0;                         // float[2048]
constexpr size_t WS_BHI_OFF = 8192;                      // u32[64*8*32*8]
constexpr size_t WS_BLO_OFF = 8192 + (size_t)NCHUNK * NT * 32 * 8 * 4;

static __device__ __forceinline__ unsigned f2bf(float f) {
  unsigned u = __float_as_uint(f);
  return (u + 0x7FFFu + ((u >> 16) & 1u)) >> 16;   // round-to-nearest-even
}
static __device__ __forceinline__ float bf2f(unsigned h) {
  return __uint_as_float(h << 16);
}

// --- prep 1: s[f] = sum_k kernel[f,k]^2 ------------------------------------
__global__ void fm_sumsq_kernel(const float* __restrict__ ker,
                                float* __restrict__ s) {
  int f = blockIdx.x * blockDim.x + threadIdx.x;
  if (f >= Fdim) return;
  const float* row = ker + (size_t)f * Kdim;
  float acc = 0.f;
  #pragma unroll 8
  for (int k = 0; k < Kdim; ++k) { float v = row[k]; acc = fmaf(v, v, acc); }
  s[f] = acc;
}

// --- prep 2: pre-swizzle kernel into WMMA B-fragment layout (hi/lo bf16) ---
// B 32x16 bf16 layout (wave32): lanes 0-15 hold K={0..7,16..23}, lanes 16-31 K={8..15,24..31};
// dword j holds K = {base+2j+(j>=4?8:0) (lo16), +1 (hi16)} for column N = lane&15.
// Storage: dword index = ((c*NT + t)*32 + lane)*8 + j   (lane-contiguous 32B).
__global__ void fm_pack_kernel(const float* __restrict__ ker,
                               unsigned* __restrict__ bhi,
                               unsigned* __restrict__ blo) {
  int id = blockIdx.x * blockDim.x + threadIdx.x;
  if (id >= NCHUNK * NT * 32 * 8) return;
  int j = id & 7;
  int l = (id >> 3) & 31;
  int t = (id >> 8) & (NT - 1);
  int c = id >> 11;
  int k0  = (l >> 4) * 8 + 2 * j + ((j >> 2) << 3);   // K within chunk
  int f   = c * CH + k0;
  int col = t * 16 + (l & 15);
  float e0 = ker[(size_t)f * Kdim + col];
  float e1 = ker[(size_t)(f + 1) * Kdim + col];
  unsigned h0 = f2bf(e0), h1 = f2bf(e1);
  float r0 = e0 - bf2f(h0), r1 = e1 - bf2f(h1);
  bhi[id] = h0 | (h1 << 16);
  blo[id] = f2bf(r0) | (f2bf(r1) << 16);
}

// --- main: per-wave 16-row stripe, 8 C tiles, F swept in chunks of 32 ------
__global__ __launch_bounds__(WG_THREADS)
void fm_main_kernel(const float* __restrict__ x,
                    const float* __restrict__ wlin,
                    const float* __restrict__ blin,
                    const float* __restrict__ s,
                    const unsigned* __restrict__ bhi,
                    const unsigned* __restrict__ blo,
                    float* __restrict__ out) {
  const int lane    = threadIdx.x & 31;
  const int wave    = threadIdx.x >> 5;
  const int rowBase = blockIdx.x * ROWS_PER_WG + wave * 16;
  const int row     = rowBase + (lane & 15);
  const int kb      = (lane >> 4) * 8;      // A layout: lane<16 -> K{0..7,16..23}

  v8f acc[NT];
  #pragma unroll
  for (int t = 0; t < NT; ++t)
    #pragma unroll
    for (int e = 0; e < 8; ++e) acc[t][e] = 0.f;

  float plin = 0.f;   // x . w_lin   (per-lane partial of one row)
  float pq   = 0.f;   // x^2 . s

  const float* xr = x + (size_t)row * Fdim;

  union AV { v16bf v; unsigned u[8]; };
  union BV { v16bf v; uint4 q[2]; };

  for (int c = 0; c < NCHUNK; ++c) {
    const int f0 = c * CH + kb;

    // stream 16 x values: [f0..f0+7] and [f0+16..f0+23]
    float4 a0 = *reinterpret_cast<const float4*>(xr + f0);
    float4 a1 = *reinterpret_cast<const float4*>(xr + f0 + 4);
    float4 a2 = *reinterpret_cast<const float4*>(xr + f0 + 16);
    float4 a3 = *reinterpret_cast<const float4*>(xr + f0 + 20);
    float xv[16] = {a0.x, a0.y, a0.z, a0.w, a1.x, a1.y, a1.z, a1.w,
                    a2.x, a2.y, a2.z, a2.w, a3.x, a3.y, a3.z, a3.w};

    if (c + 1 < NCHUNK) __builtin_prefetch(xr + f0 + CH, 0, 0);

    // linear + q partials from the same registers (w_lin, s are L2-resident)
    float4 w0 = *reinterpret_cast<const float4*>(wlin + f0);
    float4 w1 = *reinterpret_cast<const float4*>(wlin + f0 + 4);
    float4 w2 = *reinterpret_cast<const float4*>(wlin + f0 + 16);
    float4 w3 = *reinterpret_cast<const float4*>(wlin + f0 + 20);
    float4 s0 = *reinterpret_cast<const float4*>(s + f0);
    float4 s1 = *reinterpret_cast<const float4*>(s + f0 + 4);
    float4 s2 = *reinterpret_cast<const float4*>(s + f0 + 16);
    float4 s3 = *reinterpret_cast<const float4*>(s + f0 + 20);
    float wv[16] = {w0.x, w0.y, w0.z, w0.w, w1.x, w1.y, w1.z, w1.w,
                    w2.x, w2.y, w2.z, w2.w, w3.x, w3.y, w3.z, w3.w};
    float sv[16] = {s0.x, s0.y, s0.z, s0.w, s1.x, s1.y, s1.z, s1.w,
                    s2.x, s2.y, s2.z, s2.w, s3.x, s3.y, s3.z, s3.w};
    #pragma unroll
    for (int i = 0; i < 16; ++i) {
      plin = fmaf(xv[i], wv[i], plin);
      pq   = fmaf(xv[i] * xv[i], sv[i], pq);
    }

    // split x into bf16 hi/lo A-fragments (dword j = elements {2j, 2j+1})
    AV Ahi, Alo;
    #pragma unroll
    for (int j = 0; j < 8; ++j) {
      float e0 = xv[2 * j], e1 = xv[2 * j + 1];
      unsigned h0 = f2bf(e0), h1 = f2bf(e1);
      float r0 = e0 - bf2f(h0), r1 = e1 - bf2f(h1);
      Ahi.u[j] = h0 | (h1 << 16);
      Alo.u[j] = f2bf(r0) | (f2bf(r1) << 16);
    }

    #pragma unroll
    for (int t = 0; t < NT; ++t) {
      const uint4* ph =
          reinterpret_cast<const uint4*>(bhi + ((size_t)(c * NT + t) << 8) + lane * 8);
      const uint4* pl =
          reinterpret_cast<const uint4*>(blo + ((size_t)(c * NT + t) << 8) + lane * 8);
      BV Bh, Bl;
      Bh.q[0] = ph[0]; Bh.q[1] = ph[1];
      Bl.q[0] = pl[0]; Bl.q[1] = pl[1];

      acc[t] = __builtin_amdgcn_wmma_f32_16x16x32_bf16(
          false, Ahi.v, false, Bh.v, (short)0, acc[t], false, false);
      acc[t] = __builtin_amdgcn_wmma_f32_16x16x32_bf16(
          false, Ahi.v, false, Bl.v, (short)0, acc[t], false, false);
      acc[t] = __builtin_amdgcn_wmma_f32_16x16x32_bf16(
          false, Alo.v, false, Bh.v, (short)0, acc[t], false, false);
    }
  }

  // per-row sum of squares of y. C layout: VGPR g, lanes 0-15 -> row g,
  // lanes 16-31 -> row g+8; columns striped across lanes within the half.
  float ssum[8];
  #pragma unroll
  for (int g = 0; g < 8; ++g) {
    float v = 0.f;
    #pragma unroll
    for (int t = 0; t < NT; ++t) { float cv = acc[t][g]; v = fmaf(cv, cv, v); }
    #pragma unroll
    for (int m = 1; m <= 8; m <<= 1) v += __shfl_xor(v, m, 32);  // within half
    ssum[g] = v;
  }
  plin += __shfl_xor(plin, 16, 32);   // combine the two lanes of each row
  pq   += __shfl_xor(pq,   16, 32);

  const int g = lane & 7;
  float myss = ssum[0];
  #pragma unroll
  for (int gg = 1; gg < 8; ++gg) if (g == gg) myss = ssum[gg];

  // lanes 0..7 write rows 0..7 (half 0), lanes 24..31 write rows 8..15 (half 1)
  if (lane < 8 || lane >= 24) {
    int orow = rowBase + ((lane < 8) ? lane : (lane - 16));
    float cross = 0.5f * (myss - pq) * (1.0f / (float)Kdim);
    float z = plin + blin[0] + cross;
    out[orow] = 1.0f / (1.0f + __expf(-z));
  }
}

// ---------------------------------------------------------------------------
extern "C" void kernel_launch(void* const* d_in, const int* in_sizes, int n_in,
                              void* d_out, int out_size, void* d_ws, size_t ws_size,
                              hipStream_t stream) {
  const float* x    = (const float*)d_in[0];
  const float* ker  = (const float*)d_in[1];
  const float* wlin = (const float*)d_in[2];
  const float* blin = (const float*)d_in[3];
  float* out = (float*)d_out;

  char* ws = (char*)d_ws;
  float*    s   = (float*)(ws + WS_S_OFF);
  unsigned* bhi = (unsigned*)(ws + WS_BHI_OFF);
  unsigned* blo = (unsigned*)(ws + WS_BLO_OFF);

  fm_sumsq_kernel<<<(Fdim + 255) / 256, 256, 0, stream>>>(ker, s);
  fm_pack_kernel<<<(NCHUNK * NT * 32 * 8) / 256, 256, 0, stream>>>(ker, bhi, blo);
  fm_main_kernel<<<Bn / ROWS_PER_WG, WG_THREADS, 0, stream>>>(
      x, wlin, blin, s, bhi, blo, out);
}
